// Attention_77919296684050
// MI455X (gfx1250) — compile-verified, matrix operand order checked by reading, not run on hardware
//
#include <hip/hip_runtime.h>
#include <hip/hip_bf16.h>
#include <math.h>

// Problem sizes (fixed by the reference)
#define B_  32
#define S_  2048
#define QD_ 1024
#define VD_ 1024
#define AD_ 1024

typedef __bf16 bf16;
typedef bf16  v16bf __attribute__((ext_vector_type(16)));
typedef bf16  v8bf  __attribute__((ext_vector_type(8)));
typedef float v8f   __attribute__((ext_vector_type(8)));
typedef unsigned int u32x4 __attribute__((ext_vector_type(4)));
typedef int   i32x8 __attribute__((ext_vector_type(8)));
typedef int   i32x4 __attribute__((ext_vector_type(4)));

__device__ __forceinline__ bf16 f2bf(float f) { return (bf16)f; }

// Fast tanh: hardware V_TANH_F32 when the builtin exists, otherwise a
// branch-free 1 - 2/(e^{2x}+1) using V_EXP_F32 + V_RCP_F32.
__device__ __forceinline__ float fast_tanh(float x) {
#if __has_builtin(__builtin_amdgcn_tanhf)
    return __builtin_amdgcn_tanhf(x);
#else
    float e = __expf(2.0f * x);
    return 1.0f - 2.0f * __builtin_amdgcn_rcpf(e + 1.0f);
#endif
}

// ---------------------------------------------------------------------------
// Kernel 0: fp32 -> bf16 bulk convert (context and Wh). 8 elems / thread.
// ---------------------------------------------------------------------------
__global__ void cvt_bf16_kernel(const float* __restrict__ src,
                                bf16* __restrict__ dst, long long n)
{
    long long i = ((long long)blockIdx.x * blockDim.x + threadIdx.x) * 8;
    if (i + 8 > n) return;
    float4 a = *(const float4*)(src + i);
    float4 b = *(const float4*)(src + i + 4);
    v8bf o;
    o[0] = f2bf(a.x); o[1] = f2bf(a.y); o[2] = f2bf(a.z); o[3] = f2bf(a.w);
    o[4] = f2bf(b.x); o[5] = f2bf(b.y); o[6] = f2bf(b.z); o[7] = f2bf(b.w);
    *(v8bf*)(dst + i) = o;
}

// ---------------------------------------------------------------------------
// Kernel 1: output_p[b,a] = sum_q output[b,q] * Ws_w[a,q]   (32x1024, K=1024)
// One wave per 16x16 tile, WMMA bf16, fp32 accumulate.
// ---------------------------------------------------------------------------
__global__ __launch_bounds__(32) void outp_kernel(
    const float* __restrict__ output,
    const float* __restrict__ Ws_w,
    float* __restrict__ outp)
{
    const int at = blockIdx.x & 63;   // 64 N tiles (A dim)
    const int bt = blockIdx.x >> 6;   // 2 M tiles  (B dim)
    const int b0 = bt * 16, a0 = at * 16;
    const int lane = threadIdx.x & 31;
    const int mn   = lane & 15;
    const int koff = (lane >> 4) * 8;   // ISA 16-bit A/B layout: K 0-7/16-23 vs 8-15/24-31

    v8f acc = {};
    for (int kk = 0; kk < QD_; kk += 32) {
        const float* ap = output + (size_t)(b0 + mn) * QD_ + kk + koff;
        const float* bp = Ws_w   + (size_t)(a0 + mn) * QD_ + kk + koff;
        v16bf af, bfv;
#pragma unroll
        for (int j = 0; j < 8; ++j) {
            af[j]      = f2bf(ap[j]);
            af[8 + j]  = f2bf(ap[16 + j]);
            bfv[j]     = f2bf(bp[j]);
            bfv[8 + j] = f2bf(bp[16 + j]);
        }
        acc = __builtin_amdgcn_wmma_f32_16x16x32_bf16(false, af, false, bfv,
                                                      (short)0, acc, false, false);
    }
    // C layout: VGPR r, lanes 0-15 -> row r, lanes 16-31 -> row r+8; col = lane&15
    const int col = a0 + (lane & 15);
    const int rb  = b0 + (lane >> 4) * 8;
#pragma unroll
    for (int r = 0; r < 8; ++r) outp[(size_t)(rb + r) * AD_ + col] = acc[r];
}

// ---------------------------------------------------------------------------
// Kernel 2 (main): fused  scores[b,s] = v . tanh(Wh@c + bias + output_p)
// Workgroup = 8 waves, tile = 32 s-rows x 1024 a-cols, K = 1024 (V dim).
// Context tile (32x1024 bf16 = 64KB) staged into LDS by the Tensor Data Mover
// with descriptor padding (4 DWORDs / 256 DWORDs) to spread LDS banks.
// Each wave owns 128 a-columns: acc = 2 Mtiles x 8 Ntiles x v8f.
// ---------------------------------------------------------------------------

// LDS index for bf16 element k, accounting for TDM pad (4 DWORDs per 256)
__device__ __forceinline__ int lidx(int k) {
    int d = k >> 1;                 // DWORD index
    d += (d >> 8) << 2;             // +4 DWORDs after each 256-DWORD chunk
    return (d << 1) | (k & 1);
}

__global__ __launch_bounds__(256) void scores_kernel(
    const bf16* __restrict__ ctx_bf,
    const bf16* __restrict__ wh_bf,
    const float* __restrict__ Wh_b,
    const float* __restrict__ v_w,
    const float* __restrict__ outp,
    float* __restrict__ scores)
{
    __shared__ bf16  tileA[33280];   // 32*1024 bf16 + TDM padding
    __shared__ float s_red[32];

    const int b  = blockIdx.x >> 6;   // S/32 = 64 s-tiles per batch
    const int st = blockIdx.x & 63;
    const int s0 = st * 32;
    const int tid  = threadIdx.x;
    const int wave = tid >> 5;
    const int lane = tid & 31;

    if (tid < 32) s_red[tid] = 0.0f;

    // --- TDM: one descriptor per workgroup, issued by wave 0 -------------
    if (wave == 0) {
        unsigned long long ga =
            (unsigned long long)(const void*)(ctx_bf + ((size_t)b * S_ + s0) * VD_);
        unsigned int ldsoff = (unsigned int)(unsigned long long)&tileA[0];

        u32x4 g0;
        g0[0] = 1u;                                   // count=1, gather off
        g0[1] = ldsoff;                               // lds_addr
        g0[2] = (unsigned int)(ga & 0xFFFFFFFFull);   // global_addr[31:0]
        g0[3] = (unsigned int)((ga >> 32) & 0x01FFFFFFull) | 0x80000000u; // addr hi | type=2

        i32x8 g1;
        // data_size=1 (2B) | pad_enable | pad_interval=7 (256 DW) | pad_amount=3 (4 DW)
        g1[0] = (int)0x07D10000;
        g1[1] = (int)(0x8000u << 16);   // tensor_dim0 = 32768 (low16 in [63:48])
        g1[2] = (int)(1u << 16);        // tensor_dim0 hi=0 | tensor_dim1 = 1
        g1[3] = (int)(0x8000u << 16);   // tensor_dim1 hi=0 | tile_dim0 = 32768
        g1[4] = 1;                      // tile_dim1 = 1, tile_dim2 = 0
        g1[5] = 0x8000;                 // tensor_dim0_stride = 32768
        g1[6] = 0;
        g1[7] = 0;

        i32x4 g2 = {0, 0, 0, 0};
        i32x4 g3 = {0, 0, 0, 0};
        i32x8 g4 = {0, 0, 0, 0, 0, 0, 0, 0};   // unused trailing group (6-arg form)
        __builtin_amdgcn_tensor_load_to_lds(g0, g1, g2, g3, g4, 0);
        __builtin_amdgcn_s_wait_tensorcnt((short)0);
    }
    __syncthreads();

    const int a0w  = wave * 128;        // this wave's 8 N-tiles
    const int mn   = lane & 15;
    const int koff = (lane >> 4) * 8;

    v8f acc[2][8];
#pragma unroll
    for (int mt = 0; mt < 2; ++mt)
#pragma unroll
        for (int t = 0; t < 8; ++t) acc[mt][t] = (v8f){};

    for (int kk = 0; kk < VD_; kk += 32) {
        // A fragments from LDS (ds_load_b128 x2 per Mtile)
        v16bf af[2];
#pragma unroll
        for (int mt = 0; mt < 2; ++mt) {
            const int row = mt * 16 + mn;
            v8bf lo = *(const v8bf*)&tileA[lidx(row * VD_ + kk + koff)];
            v8bf hi = *(const v8bf*)&tileA[lidx(row * VD_ + kk + 16 + koff)];
            af[mt] = __builtin_shufflevector(lo, hi,
                       0,1,2,3,4,5,6,7,8,9,10,11,12,13,14,15);
        }
        // B fragments straight from L2-resident bf16 Wh (global_load_b128 x2)
#pragma unroll
        for (int t = 0; t < 8; ++t) {
            const bf16* bp = wh_bf + (size_t)(a0w + t * 16 + mn) * VD_ + kk + koff;
            v8bf blo = *(const v8bf*)bp;
            v8bf bhi = *(const v8bf*)(bp + 16);
            v16bf bfv = __builtin_shufflevector(blo, bhi,
                          0,1,2,3,4,5,6,7,8,9,10,11,12,13,14,15);
#pragma unroll
            for (int mt = 0; mt < 2; ++mt)
                acc[mt][t] = __builtin_amdgcn_wmma_f32_16x16x32_bf16(
                    false, af[mt], false, bfv, (short)0, acc[mt][t], false, false);
        }
    }

    // Epilogue: bias + output_p, hardware tanh, dot with v, ds_add reduce
    const int rbase = (lane >> 4) * 8;
#pragma unroll
    for (int mt = 0; mt < 2; ++mt) {
        float part[8] = {0, 0, 0, 0, 0, 0, 0, 0};
#pragma unroll
        for (int t = 0; t < 8; ++t) {
            const int a = a0w + t * 16 + mn;
            const float add = Wh_b[a] + outp[(size_t)b * AD_ + a];
            const float vw  = v_w[a];
#pragma unroll
            for (int r = 0; r < 8; ++r)
                part[r] += fast_tanh(acc[mt][t][r] + add) * vw;
        }
#pragma unroll
        for (int r = 0; r < 8; ++r)
            atomicAdd(&s_red[mt * 16 + rbase + r], part[r]);  // ds_add_f32
    }
    __syncthreads();
    if (tid < 32) scores[(size_t)b * S_ + s0 + tid] = s_red[tid];
}

// ---------------------------------------------------------------------------
// Kernel 3: masked softmax over S, then weighted = p_attn @ context (bf16)
// One block (16 waves) per batch row.
// ---------------------------------------------------------------------------
__global__ __launch_bounds__(512) void softmax_weighted_kernel(
    const float* __restrict__ scores,
    const int* __restrict__ mask,
    const bf16* __restrict__ ctx_bf,
    float* __restrict__ out_w,
    float* __restrict__ out_p)
{
    __shared__ float p_sh[S_];
    __shared__ float red[16];
    const int b = blockIdx.x;
    const int tid = threadIdx.x;
    const int lane = tid & 31, wid = tid >> 5;

    float lmax = -3.0e38f;
    for (int s = tid; s < S_; s += 512) {
        float sc = (mask[b * S_ + s] == 0) ? -1.0e9f : scores[(size_t)b * S_ + s];
        p_sh[s] = sc;
        lmax = fmaxf(lmax, sc);
    }
#pragma unroll
    for (int o = 16; o; o >>= 1) lmax = fmaxf(lmax, __shfl_xor(lmax, o, 32));
    if (lane == 0) red[wid] = lmax;
    __syncthreads();
    if (tid == 0) {
        float m = red[0];
        for (int i = 1; i < 16; ++i) m = fmaxf(m, red[i]);
        red[0] = m;
    }
    __syncthreads();
    const float gmax = red[0];
    __syncthreads();

    float lsum = 0.0f;
    for (int s = tid; s < S_; s += 512) {
        float e = __expf(p_sh[s] - gmax);
        p_sh[s] = e;
        lsum += e;
    }
#pragma unroll
    for (int o = 16; o; o >>= 1) lsum += __shfl_xor(lsum, o, 32);
    if (lane == 0) red[wid] = lsum;
    __syncthreads();
    if (tid == 0) {
        float t = 0.0f;
        for (int i = 0; i < 16; ++i) t += red[i];
        red[0] = t;
    }
    __syncthreads();
    const float inv = 1.0f / red[0];
    for (int s = tid; s < S_; s += 512) {
        float p = p_sh[s] * inv;
        p_sh[s] = p;
        out_p[(size_t)b * S_ + s] = p;
    }
    __syncthreads();

    // weighted[b,v] = sum_s p[s] * context[b,s,v]  (coalesced across v)
    for (int v = tid; v < VD_; v += 512) {
        float accv = 0.0f;
        const bf16* cp = ctx_bf + (size_t)b * S_ * VD_ + v;
        for (int s = 0; s < S_; ++s) accv += p_sh[s] * (float)cp[(size_t)s * VD_];
        out_w[(size_t)b * VD_ + v] = accv;
    }
}

// ---------------------------------------------------------------------------
extern "C" void kernel_launch(void* const* d_in, const int* in_sizes, int n_in,
                              void* d_out, int out_size, void* d_ws, size_t ws_size,
                              hipStream_t stream)
{
    (void)in_sizes; (void)n_in; (void)out_size; (void)ws_size;
    const float* output  = (const float*)d_in[0];
    const float* context = (const float*)d_in[1];
    const int*   mask    = (const int*)d_in[2];
    const float* Wh_w    = (const float*)d_in[3];
    const float* Wh_b    = (const float*)d_in[4];
    const float* Ws_w    = (const float*)d_in[5];
    const float* v_w     = (const float*)d_in[6];

    // Workspace layout
    char* ws = (char*)d_ws;
    bf16*  ctx_bf = (bf16*)ws;                                       // 128 MB
    bf16*  wh_bf  = (bf16*)(ws + (size_t)B_ * S_ * VD_ * 2);         //   2 MB
    float* outp   = (float*)((char*)wh_bf + (size_t)AD_ * VD_ * 2);  // 128 KB
    float* scores = (float*)((char*)outp + (size_t)B_ * AD_ * 4);    // 256 KB

    float* out_w = (float*)d_out;            // weighted [B, VD]
    float* out_p = out_w + (size_t)B_ * VD_; // p_attn   [B, 1, S]

    {   // context fp32 -> bf16  (read once, both passes then read bf16)
        long long n = (long long)B_ * S_ * VD_;
        cvt_bf16_kernel<<<(unsigned)(n / (8 * 256)), 256, 0, stream>>>(context, ctx_bf, n);
    }
    {   // Wh fp32 -> bf16 (L2-resident weight panel)
        long long n = (long long)AD_ * VD_;
        cvt_bf16_kernel<<<(unsigned)(n / (8 * 256)), 256, 0, stream>>>(Wh_w, wh_bf, n);
    }
    outp_kernel<<<2 * 64, 32, 0, stream>>>(output, Ws_w, outp);
    scores_kernel<<<B_ * (S_ / 32), 256, 0, stream>>>(ctx_bf, wh_bf, Wh_b, v_w,
                                                      outp, scores);
    softmax_weighted_kernel<<<B_, 512, 0, stream>>>(scores, mask, ctx_bf,
                                                    out_w, out_p);
}
